// ShapeGainCodebook_88510686036491
// MI455X (gfx1250) — matrix operand order repaired and checked by reading, not compiled, not run on hardware
//
#include <hip/hip_runtime.h>
#include <stdint.h>

#define DIMS        256
#define SHAPE_KK    1024
#define GAIN_KK     64
#define ROWS_PER_WG 128
#define CT_TILES    (SHAPE_KK / 16)   // 64 code tiles of 16
#define CB_STRIDE   132               // dwords per code row: 128 pairs + 4 pad (banks 0,4,...,60; 16B aligned)

typedef __attribute__((ext_vector_type(16))) __bf16   v16bf;
typedef __attribute__((ext_vector_type(8)))  float    v8f;
typedef __attribute__((ext_vector_type(8)))  uint32_t v8u;

// Split fp32 into truncated-hi bf16 + RNE-rounded residual bf16 (bit manipulation only).
__device__ __forceinline__ void splitf(float f, uint32_t& h, uint32_t& l) {
    union { float f; uint32_t u; } a; a.f = f;
    uint32_t hb = a.u & 0xFFFF0000u;
    union { uint32_t u; float f; } hf; hf.u = hb;
    float r = f - hf.f;
    union { float f; uint32_t u; } b; b.f = r;
    uint32_t ru = b.u + 0x7FFFu + ((b.u >> 16) & 1u);   // round-to-nearest-even
    h = hb >> 16;
    l = ru >> 16;
}

__device__ __forceinline__ float fatomic_add(float* p, float v) {
    return __hip_atomic_fetch_add(p, v, __ATOMIC_RELAXED, __HIP_MEMORY_SCOPE_AGENT);
}

__device__ __forceinline__ v8f wmma_bf16(v8u a, v8u b, v8f c) {
    return __builtin_amdgcn_wmma_f32_16x16x32_bf16(
        false, __builtin_bit_cast(v16bf, a),
        false, __builtin_bit_cast(v16bf, b),
        (short)0, c, false, false);
}

// Load 8 contiguous LDS dwords as one v8u (two ds_load_b128, no cross-lane shuffles).
__device__ __forceinline__ v8u ldsv8(const uint32_t* p) {
    uint4 a = *(const uint4*)p;
    uint4 b = *(const uint4*)(p + 4);
    v8u r;
    r[0] = a.x; r[1] = a.y; r[2] = a.z; r[3] = a.w;
    r[4] = b.x; r[5] = b.y; r[6] = b.z; r[7] = b.w;
    return r;
}

// ---------------------------------------------------------------------------
// Main fused kernel: distances (split-bf16 WMMA) + argmax + gain + quantize +
// scatter accumulation of codebook statistics.
// ---------------------------------------------------------------------------
__global__ __launch_bounds__(256)
void vq_main(const float* __restrict__ x, const float* __restrict__ shape_cb,
             const float* __restrict__ gain_cb, float* __restrict__ outQ,
             float* __restrict__ wsShapeAcc, float* __restrict__ wsShapeCnt,
             float* __restrict__ wsGainSum, float* __restrict__ wsGainCnt,
             int nRows)
{
    // Separate hi/lo planes: each B fragment = 8 contiguous dwords.
    __shared__ uint32_t cbHi[2][16 * CB_STRIDE];
    __shared__ uint32_t cbLo[2][16 * CB_STRIDE];
    __shared__ float gcbS[GAIN_KK];
    __shared__ int   bestIdxS[ROWS_PER_WG];
    __shared__ float rowGainS[ROWS_PER_WG];

    const int tid  = threadIdx.x;
    const int lane = tid & 31;
    const int wave = tid >> 5;
    const int half = lane >> 4;     // WMMA lane half
    const int hl   = lane & 15;
    const int rowBase0 = blockIdx.x * ROWS_PER_WG;

    if (tid < GAIN_KK) gcbS[tid] = gain_cb[tid];

    // ---- A fragments: 16 rows per wave, full D=256, split hi/lo bf16, kept as v8u vectors
    v8u aH[8], aL[8];
    {
        int r = rowBase0 + wave * 16 + hl;
        if (r > nRows - 1) r = nRows - 1;
        const float* xr = x + (size_t)r * DIMS;
        #pragma unroll
        for (int dc = 0; dc < 8; ++dc) {
            #pragma unroll
            for (int j = 0; j < 8; ++j) {
                // A k order used here, same as B: VGPR j pairs contiguous in K; lanes 16-31 shift K by +16
                int k0 = dc * 32 + 2 * j + 16 * half;
                float2 v = *(const float2*)(xr + k0);
                uint32_t h0, l0, h1, l1;
                splitf(v.x, h0, l0); splitf(v.y, h1, l1);
                aH[dc][j] = h0 | (h1 << 16);
                aL[dc][j] = l0 | (l1 << 16);
            }
        }
    }

    // ---- codebook tile staging (16 codes x 256 dims per tile; 256 threads x 16 dims)
    const int codeL = tid >> 4;          // code within tile
    const int dim0  = (tid & 15) * 16;   // 16 dims handled by this thread
    const int pair0 = (tid & 15) * 8;    // 8 packed dwords per plane
    float4 pre[4];

    auto fetchTile = [&](int ct) {
        const float4* s = (const float4*)(shape_cb + (size_t)(ct * 16 + codeL) * DIMS + dim0);
        pre[0] = s[0]; pre[1] = s[1]; pre[2] = s[2]; pre[3] = s[3];
    };
    auto storeTile = [&](int buf) {
        const float* pf = (const float*)pre;
        uint32_t hw[8], lw[8];
        #pragma unroll
        for (int p = 0; p < 8; ++p) {
            uint32_t h0, l0, h1, l1;
            splitf(pf[2 * p], h0, l0); splitf(pf[2 * p + 1], h1, l1);
            hw[p] = h0 | (h1 << 16);
            lw[p] = l0 | (l1 << 16);
        }
        uint32_t* dh = &cbHi[buf][codeL * CB_STRIDE + pair0];
        uint32_t* dl = &cbLo[buf][codeL * CB_STRIDE + pair0];
        *(uint4*)(dh)     = make_uint4(hw[0], hw[1], hw[2], hw[3]);
        *(uint4*)(dh + 4) = make_uint4(hw[4], hw[5], hw[6], hw[7]);
        *(uint4*)(dl)     = make_uint4(lw[0], lw[1], lw[2], lw[3]);
        *(uint4*)(dl + 4) = make_uint4(lw[4], lw[5], lw[6], lw[7]);
    };

    float bestV[8]; int bestI[8];
    #pragma unroll
    for (int r = 0; r < 8; ++r) { bestV[r] = -3.402823466e38f; bestI[r] = 0; }

    fetchTile(0);
    storeTile(0);
    __syncthreads();
    fetchTile(1);

    // ---- main loop: 64 code tiles, 3-term split-bf16 WMMA, fused running argmax.
    // Two d-chunks per stage: all four B fragments are loaded before the first
    // WMMA of the stage, so loads for chunk dc+1 stay outstanding underneath the
    // 3-WMMA group of chunk dc (counter-based s_wait_dscnt overlap).
    for (int ct = 0; ct < CT_TILES; ++ct) {
        const int buf = ct & 1;
        const uint32_t* hiP = &cbHi[buf][hl * CB_STRIDE + 8 * half];
        const uint32_t* loP = &cbLo[buf][hl * CB_STRIDE + 8 * half];
        v8f acc = {0.f, 0.f, 0.f, 0.f, 0.f, 0.f, 0.f, 0.f};

        #pragma unroll
        for (int dc = 0; dc < 8; dc += 2) {
            v8u bh0 = ldsv8(hiP + dc * 16);
            v8u bl0 = ldsv8(loP + dc * 16);
            v8u bh1 = ldsv8(hiP + (dc + 1) * 16);
            v8u bl1 = ldsv8(loP + (dc + 1) * 16);
            acc = wmma_bf16(aH[dc], bh0, acc);       // hi*hi
            acc = wmma_bf16(aL[dc], bh0, acc);       // lo*hi
            acc = wmma_bf16(aH[dc], bl0, acc);       // hi*lo -> ~fp32 fidelity
            acc = wmma_bf16(aH[dc + 1], bh1, acc);
            acc = wmma_bf16(aL[dc + 1], bh1, acc);
            acc = wmma_bf16(aH[dc + 1], bl1, acc);
        }

        #pragma unroll
        for (int r = 0; r < 8; ++r) {                // C layout: VGPR r -> row r+8*half, col = hl
            float v = acc[r];
            if (v > bestV[r]) { bestV[r] = v; bestI[r] = ct * 16 + hl; }
        }
        if (ct + 1 < CT_TILES) {
            storeTile(buf ^ 1);
            if (ct + 2 < CT_TILES) fetchTile(ct + 2);
        }
        __syncthreads();
    }

    // ---- reduce (max, first-argmax) across the 16 lanes holding each row
    #pragma unroll
    for (int m = 1; m <= 8; m <<= 1) {
        #pragma unroll
        for (int r = 0; r < 8; ++r) {
            float ov = __shfl_xor(bestV[r], m, 32);
            int   oi = __shfl_xor(bestI[r], m, 32);
            if (ov > bestV[r] || (ov == bestV[r] && oi < bestI[r])) {
                bestV[r] = ov; bestI[r] = oi;
            }
        }
    }
    if (hl == 0) {
        #pragma unroll
        for (int r = 0; r < 8; ++r)
            bestIdxS[wave * 16 + 8 * half + r] = bestI[r];
    }
    __syncthreads();

    // ---- phase 2: exact fp32 gain, log, 64-entry gain argmin, gain stats
    {
        const int row = tid >> 1, sub = tid & 1;
        const int grow = rowBase0 + row;
        const bool valid = grow < nRows;
        const int k = bestIdxS[row];
        const float* xr = x + (size_t)(valid ? grow : 0) * DIMS;
        const float* cr = shape_cb + (size_t)k * DIMS;
        float s = 0.f;
        for (int d = sub * 128; d < sub * 128 + 128; d += 4) {
            float4 a = *(const float4*)(xr + d);
            float4 b = *(const float4*)(cr + d);
            s += a.x * b.x + a.y * b.y + a.z * b.z + a.w * b.w;
        }
        s += __shfl_xor(s, 1, 32);
        if (sub == 0) {
            float gf = logf(fmaxf(s, 1e-5f));          // LOG_GAIN + clip(EPS)
            float bd = 3.402823466e38f; int bi = 0;
            #pragma unroll
            for (int j = 0; j < GAIN_KK; ++j) {        // argmax of -(gf-g)^2, first max
                float df = gf - gcbS[j];
                float d2 = df * df;
                if (d2 < bd) { bd = d2; bi = j; }
            }
            rowGainS[row] = expf(gcbS[bi]);            // exp(gain_q)
            if (valid) {
                fatomic_add(&wsGainSum[bi], gf);
                fatomic_add(&wsGainCnt[bi], 1.f);
                fatomic_add(&wsShapeCnt[k], 1.f);
            }
        }
    }
    __syncthreads();

    // ---- phase 3: quantize write + scatter x rows into per-cluster accumulator
    {
        const int d = tid;   // 256 threads == DIMS
        for (int r = 0; r < ROWS_PER_WG; ++r) {
            const int grow = rowBase0 + r;
            if (grow >= nRows) break;
            const int k = bestIdxS[r];
            const float gq = rowGainS[r];
            const float cv = shape_cb[(size_t)k * DIMS + d];
            outQ[(size_t)grow * DIMS + d] = gq * cv;
            fatomic_add(&wsShapeAcc[(size_t)k * DIMS + d], x[(size_t)grow * DIMS + d]);
        }
    }
}

// ---------------------------------------------------------------------------
// Shape codebook finalize: normalize column sums, EMA, re-L2-normalize.
// ---------------------------------------------------------------------------
__global__ __launch_bounds__(256)
void vq_shape_fin(const float* __restrict__ shape_cb, const float* __restrict__ shape_num,
                  const float* __restrict__ wsShapeAcc, const float* __restrict__ wsShapeCnt,
                  float* __restrict__ outShape, float* __restrict__ outSnum)
{
    __shared__ float red[256];
    const int k = blockIdx.x, d = threadIdx.x;
    float a = wsShapeAcc[(size_t)k * DIMS + d];
    red[d] = a * a; __syncthreads();
    for (int s = 128; s > 0; s >>= 1) { if (d < s) red[d] += red[d + s]; __syncthreads(); }
    float n1 = sqrtf(red[0]);
    __syncthreads();
    float sn = a / fmaxf(n1, 1e-5f);                          // clip(norm, EPS)
    float u = shape_cb[(size_t)k * DIMS + d] * 0.99f + sn * 0.01f;
    red[d] = u * u; __syncthreads();
    for (int s = 128; s > 0; s >>= 1) { if (d < s) red[d] += red[d + s]; __syncthreads(); }
    float n2 = fmaxf(sqrtf(red[0]), 1e-12f);                  // _l2norm clip
    outShape[(size_t)k * DIMS + d] = u / n2;
    if (d == 0) outSnum[k] = shape_num[k] * 0.99f + wsShapeCnt[k] * 0.01f;
}

__global__ void vq_gain_fin(const float* __restrict__ gain_cb, const float* __restrict__ gain_num,
                            const float* __restrict__ gsum, const float* __restrict__ gcnt,
                            float* __restrict__ outGain, float* __restrict__ outGnum)
{
    int j = threadIdx.x;
    if (j < GAIN_KK) {
        float c = gcnt[j];
        float gn = gsum[j] / fmaxf(c, 1e-5f);
        outGain[j] = gain_cb[j] * 0.99f + gn * 0.01f;
        outGnum[j] = gain_num[j] * 0.99f + c * 0.01f;
    }
}

__global__ void vq_zero(float* __restrict__ p, int n)
{
    int i = blockIdx.x * 256 + threadIdx.x;
    if (i < n) p[i] = 0.f;
}

extern "C" void kernel_launch(void* const* d_in, const int* in_sizes, int n_in,
                              void* d_out, int out_size, void* d_ws, size_t ws_size,
                              hipStream_t stream)
{
    const float* x   = (const float*)d_in[0];
    const float* scb = (const float*)d_in[1];
    const float* gcb = (const float*)d_in[2];
    const float* snm = (const float*)d_in[3];
    const float* gnm = (const float*)d_in[4];
    float* out = (float*)d_out;
    const int nRows = in_sizes[0] / DIMS;   // 48000

    float* ws         = (float*)d_ws;
    float* wsShapeAcc = ws;                             // K*D
    float* wsShapeCnt = wsShapeAcc + SHAPE_KK * DIMS;   // K
    float* wsGainSum  = wsShapeCnt + SHAPE_KK;          // GK
    float* wsGainCnt  = wsGainSum + GAIN_KK;            // GK
    const int wsFloats = SHAPE_KK * DIMS + SHAPE_KK + 2 * GAIN_KK;

    float* outQ     = out;                              // N*D
    float* outShape = outQ + (size_t)nRows * DIMS;      // K*D
    float* outGain  = outShape + SHAPE_KK * DIMS;       // GK
    float* outSnum  = outGain + GAIN_KK;                // K
    float* outGnum  = outSnum + SHAPE_KK;               // GK

    vq_zero<<<(wsFloats + 255) / 256, 256, 0, stream>>>(ws, wsFloats);
    vq_main<<<(nRows + ROWS_PER_WG - 1) / ROWS_PER_WG, 256, 0, stream>>>(
        x, scb, gcb, outQ, wsShapeAcc, wsShapeCnt, wsGainSum, wsGainCnt, nRows);
    vq_shape_fin<<<SHAPE_KK, 256, 0, stream>>>(scb, snm, wsShapeAcc, wsShapeCnt, outShape, outSnum);
    vq_gain_fin<<<1, GAIN_KK, 0, stream>>>(gcb, gnm, wsGainSum, wsGainCnt, outGain, outGnum);
}
